// PhaseScanLayer_14207751815389
// MI455X (gfx1250) — matrix operand order repaired and checked by reading, not compiled, not run on hardware
//
#include <hip/hip_runtime.h>

typedef __attribute__((ext_vector_type(16))) _Float16 v16h;
typedef __attribute__((ext_vector_type(8)))  _Float16 v8h;
typedef __attribute__((ext_vector_type(4)))  _Float16 h4;
typedef __attribute__((ext_vector_type(8)))  float    v8f;

#define B_ 8
#define L_ 2048
#define K_ 1024
#define H_ 2048          // K*exp
#define F_ 2048          // 2K
#define M_ (B_ * L_)     // 16384
#define LT 32            // L tile for scan stages
#define NT (L_ / LT)     // 64 tiles per batch

union U16 { v16h v; v8h h[2]; };

// ---- CDNA5 async global->LDS DMA (no VGPR data path, tracked by ASYNCcnt) ----
__device__ __forceinline__ void async_load_b128(unsigned lds_off, const _Float16* g) {
  asm volatile("global_load_async_to_lds_b128 %0, %1, off"
               :: "v"(lds_off), "v"(g) : "memory");
}
#define WAIT_ASYNC_4() asm volatile("s_wait_asynccnt 0x4" ::: "memory")
#define WAIT_ASYNC_0() asm volatile("s_wait_asynccnt 0x0" ::: "memory")

// ---------------- block reduction over 256 threads (8 wave32) ----------------
__device__ inline float4 blockReduce4(float4 v, float4* sred, int tid) {
  #pragma unroll
  for (int off = 16; off > 0; off >>= 1) {
    v.x += __shfl_xor(v.x, off, 32);
    v.y += __shfl_xor(v.y, off, 32);
    v.z += __shfl_xor(v.z, off, 32);
    v.w += __shfl_xor(v.w, off, 32);
  }
  int w = tid >> 5;
  if ((tid & 31) == 0) sred[w] = v;
  __syncthreads();
  if (tid == 0) {
    float4 s = sred[0];
    #pragma unroll
    for (int i = 1; i < 8; ++i) {
      float4 t = sred[i];
      s.x += t.x; s.y += t.y; s.z += t.z; s.w += t.w;
    }
    sred[8] = s;
  }
  __syncthreads();
  return sred[8];
}

// ---------------- fp32 -> f16 conversion ----------------
__global__ void f32_to_f16_kernel(const float* __restrict__ in, _Float16* __restrict__ out, int n4) {
  int i = blockIdx.x * 256 + threadIdx.x;
  if (i < n4) {
    float4 v = ((const float4*)in)[i];
    h4 h = { (_Float16)v.x, (_Float16)v.y, (_Float16)v.z, (_Float16)v.w };
    ((h4*)out)[i] = h;
  }
}

// ---------------- scan pass 1: per-(b,tile,k) partial sums over LT rows ----------------
__global__ void scan_partial_kernel(const float* __restrict__ phi, float* __restrict__ part) {
  int blk = blockIdx.x;                   // b*NT + tile
  int b = blk / NT, tile = blk % NT;
  int tid = threadIdx.x;
  const float4* p = (const float4*)(phi + ((size_t)(b * L_ + tile * LT)) * K_) + tid;
  float4 s = {0.f, 0.f, 0.f, 0.f};
  for (int l = 0; l < LT; ++l) {
    float4 v = p[(size_t)l * (K_ / 4)];
    s.x += v.x; s.y += v.y; s.z += v.z; s.w += v.w;
  }
  ((float4*)(part + (size_t)blk * K_))[tid] = s;
}

// ---------------- scan pass 2: exclusive scan across NT tiles per (b,k) ----------------
__global__ void scan_tiles_kernel(float* __restrict__ part) {
  int g = blockIdx.x * 256 + threadIdx.x;  // b*K_ + k   (8192 total)
  int b = g / K_, k = g % K_;
  size_t base = (size_t)b * NT * K_ + k;
  float run = 0.f;
  for (int t = 0; t < NT; ++t) {
    float v = part[base + (size_t)t * K_];
    part[base + (size_t)t * K_] = run;
    run += v;
  }
}

// ---------------- scan pass 3: mix + phase-norm, emit phi1 (f32) and x=[cos,sin] (f16) ----------------
__global__ __launch_bounds__(256)
void phase_stage1_kernel(const float* __restrict__ phi, const float* __restrict__ part,
                         const float* __restrict__ alphap,
                         const float* __restrict__ gamma, const float* __restrict__ beta,
                         float* __restrict__ phi1, _Float16* __restrict__ xbuf) {
  __shared__ float4 sred[9];
  int blk = blockIdx.x;
  int b = blk / NT, tile = blk % NT;
  int tid = threadIdx.x;
  int k0 = tid * 4;
  float a = 1.f / (1.f + expf(-alphap[0]));
  const float inv = 1.f / (float)K_;

  float4 g4 = ((const float4*)gamma)[tid];
  float4 b4 = ((const float4*)beta)[tid];
  float4 acc = ((const float4*)(part + (size_t)blk * K_))[tid];  // exclusive prefix

  for (int l = 0; l < LT; ++l) {
    size_t m = (size_t)b * L_ + (size_t)tile * LT + l;
    float4 p = ((const float4*)(phi + m * K_))[tid];
    acc.x += p.x; acc.y += p.y; acc.z += p.z; acc.w += p.w;   // inclusive cumsum
    float mx = p.x + a * (acc.x - p.x);
    float my = p.y + a * (acc.y - p.y);
    float mz = p.z + a * (acc.z - p.z);
    float mw = p.w + a * (acc.w - p.w);
    float cx = cosf(mx), cy = cosf(my), cz = cosf(mz), cw = cosf(mw);
    float sx = sinf(mx), sy = sinf(my), sz = sinf(mz), sw = sinf(mw);
    float4 r = { cx + cy + cz + cw, cx*cx + cy*cy + cz*cz + cw*cw,
                 sx + sy + sz + sw, sx*sx + sy*sy + sz*sz + sw*sw };
    float4 tot = blockReduce4(r, sred, tid);
    float muc = tot.x * inv, mus = tot.z * inv;
    float rc = rsqrtf(tot.y * inv - muc * muc + 1e-5f);
    float rs = rsqrtf(tot.w * inv - mus * mus + 1e-5f);
    float4 ph;
    ph.x = p.x + g4.x * atan2f((sx - mus) * rs, (cx - muc) * rc) + b4.x;
    ph.y = p.y + g4.y * atan2f((sy - mus) * rs, (cy - muc) * rc) + b4.y;
    ph.z = p.z + g4.z * atan2f((sz - mus) * rs, (cz - muc) * rc) + b4.z;
    ph.w = p.w + g4.w * atan2f((sw - mus) * rs, (cw - muc) * rc) + b4.w;
    ((float4*)(phi1 + m * K_))[tid] = ph;
    h4 xc = { (_Float16)cosf(ph.x), (_Float16)cosf(ph.y), (_Float16)cosf(ph.z), (_Float16)cosf(ph.w) };
    h4 xs = { (_Float16)sinf(ph.x), (_Float16)sinf(ph.y), (_Float16)sinf(ph.z), (_Float16)sinf(ph.w) };
    *(h4*)(xbuf + m * F_ + k0)      = xc;
    *(h4*)(xbuf + m * F_ + K_ + k0) = xs;
  }
}

// ---------------- WMMA GEMM: C[M,N] = A[M,2048] * Bw[N,2048]^T, f16 in / f32 acc ----------------
// Macro tile 128x128, 8 waves (2x4), each wave 64x32 (4x2 wmma tiles), K step 32.
// Global->LDS via async DMA (ASYNCcnt), triple-buffered LDS, 1 barrier/iter.
// EPI 0: +bias, exact GELU -> f16.   EPI 1: +bias +add (phi1) -> f32.
template <int EPI>
__global__ __launch_bounds__(256)
void wmma_gemm_kernel(const _Float16* __restrict__ A, const _Float16* __restrict__ Bw,
                      const float* __restrict__ bias, const float* __restrict__ add,
                      _Float16* __restrict__ outH, float* __restrict__ outF) {
  constexpr int KD = 2048;
  constexpr int KT = KD / 32;
  constexpr unsigned TILEH = 128 * 40;          // halfs per buffer (rows padded: 80B, 16B aligned)
  __shared__ _Float16 As[3 * TILEH];
  __shared__ _Float16 Bs[3 * TILEH];

  int tid = threadIdx.x;
  int lane = tid & 31, wid = tid >> 5;
  int wm = wid >> 2, wn = wid & 3;
  int mlane = lane & 15, kg = lane >> 4;
  int rowBase = blockIdx.y * 128;
  int colBase = blockIdx.x * 128;
  int N = gridDim.x * 128;

  int seg_row[2], seg_col[2];
  #pragma unroll
  for (int s = 0; s < 2; ++s) {
    int seg = tid + s * 256;
    seg_row[s] = seg >> 2;          // 0..127
    seg_col[s] = (seg & 3) * 8;     // 0/8/16/24 halfs
  }
  unsigned asBase = (unsigned)(uintptr_t)(void*)As;   // flat->LDS: low 32 bits
  unsigned bsBase = (unsigned)(uintptr_t)(void*)Bs;

  // issue the 4 async b128 DMAs (2 A segs + 2 B segs) for K-slab kt into buffer buf
  auto issue = [&](int kt, int buf) {
    int kBase = kt * 32;
    unsigned bufOff = (unsigned)buf * (TILEH * 2u);   // bytes
    #pragma unroll
    for (int s = 0; s < 2; ++s) {
      unsigned loff = (unsigned)(seg_row[s] * 40 + seg_col[s]) * 2u;
      async_load_b128(asBase + bufOff + loff,
                      A  + (size_t)(rowBase + seg_row[s]) * KD + kBase + seg_col[s]);
      async_load_b128(bsBase + bufOff + loff,
                      Bw + (size_t)(colBase + seg_row[s]) * KD + kBase + seg_col[s]);
    }
  };

  v8f acc[4][2];
  #pragma unroll
  for (int i = 0; i < 4; ++i)
    #pragma unroll
    for (int j = 0; j < 2; ++j)
      #pragma unroll
      for (int e = 0; e < 8; ++e) acc[i][j][e] = 0.0f;

  issue(0, 0);                       // 4 async ops in flight
  issue(1, 1);                       // 8 async ops in flight

  for (int kt = 0; kt < KT; ++kt) {
    int buf = kt % 3;
    if (kt + 1 < KT) { WAIT_ASYNC_4(); }   // own 4 DMAs of tile kt done (in-order)
    else             { WAIT_ASYNC_0(); }
    __syncthreads();                        // all waves' DMAs for tile kt visible;
                                            // also: everyone done reading buf (kt+2)%3 at kt-1
    if (kt + 2 < KT) issue(kt + 2, (kt + 2) % 3);   // overlaps with compute below

    const _Float16* as = As + (unsigned)buf * TILEH;
    const _Float16* bs = Bs + (unsigned)buf * TILEH;
    U16 afr[4], bfr[2];
    #pragma unroll
    for (int i = 0; i < 4; ++i) {
      const _Float16* base = as + (wm * 64 + i * 16 + mlane) * 40;
      afr[i].h[0] = *(const v8h*)(base + kg * 8);        // K = {0..7}/{8..15}
      afr[i].h[1] = *(const v8h*)(base + 16 + kg * 8);   // K = {16..23}/{24..31}
    }
    #pragma unroll
    for (int j = 0; j < 2; ++j) {
      const _Float16* base = bs + (wn * 32 + j * 16 + mlane) * 40;
      bfr[j].h[0] = *(const v8h*)(base + kg * 8);
      bfr[j].h[1] = *(const v8h*)(base + 16 + kg * 8);
    }
    #pragma unroll
    for (int i = 0; i < 4; ++i)
      #pragma unroll
      for (int j = 0; j < 2; ++j)
        acc[i][j] = __builtin_amdgcn_wmma_f32_16x16x32_f16(
            false, afr[i].v, false, bfr[j].v, (short)0, acc[i][j], false, false);
  }

  #pragma unroll
  for (int i = 0; i < 4; ++i) {
    #pragma unroll
    for (int j = 0; j < 2; ++j) {
      #pragma unroll
      for (int r = 0; r < 8; ++r) {
        int row = rowBase + wm * 64 + i * 16 + kg * 8 + r;
        int col = colBase + wn * 32 + j * 16 + mlane;
        float v = acc[i][j][r] + bias[col];
        if (EPI == 0) {
          v = 0.5f * v * (1.0f + erff(v * 0.70710678118654752f));   // exact GELU
          outH[(size_t)row * N + col] = (_Float16)v;
        } else {
          outF[(size_t)row * N + col] = v + add[(size_t)row * N + col];
        }
      }
    }
  }
}

// ---------------- final phase-norm over K per row ----------------
__global__ __launch_bounds__(256)
void phase_final_kernel(const float* __restrict__ z, const float* __restrict__ gamma,
                        const float* __restrict__ beta, float* __restrict__ out) {
  __shared__ float4 sred[9];
  int m = blockIdx.x;
  int tid = threadIdx.x;
  float4 zv = ((const float4*)(z + (size_t)m * K_))[tid];
  float cx = cosf(zv.x), cy = cosf(zv.y), cz = cosf(zv.z), cw = cosf(zv.w);
  float sx = sinf(zv.x), sy = sinf(zv.y), sz = sinf(zv.z), sw = sinf(zv.w);
  float4 r = { cx + cy + cz + cw, cx*cx + cy*cy + cz*cz + cw*cw,
               sx + sy + sz + sw, sx*sx + sy*sy + sz*sz + sw*sw };
  float4 tot = blockReduce4(r, sred, tid);
  const float inv = 1.f / (float)K_;
  float muc = tot.x * inv, mus = tot.z * inv;
  float rc = rsqrtf(tot.y * inv - muc * muc + 1e-5f);
  float rs = rsqrtf(tot.w * inv - mus * mus + 1e-5f);
  float4 g4 = ((const float4*)gamma)[tid];
  float4 b4 = ((const float4*)beta)[tid];
  float4 o;
  o.x = g4.x * atan2f((sx - mus) * rs, (cx - muc) * rc) + b4.x;
  o.y = g4.y * atan2f((sy - mus) * rs, (cy - muc) * rc) + b4.y;
  o.z = g4.z * atan2f((sz - mus) * rs, (cz - muc) * rc) + b4.z;
  o.w = g4.w * atan2f((sw - mus) * rs, (cw - muc) * rc) + b4.w;
  ((float4*)(out + (size_t)m * K_))[tid] = o;
}

extern "C" void kernel_launch(void* const* d_in, const int* in_sizes, int n_in,
                              void* d_out, int out_size, void* d_ws, size_t ws_size,
                              hipStream_t stream) {
  const float* phi    = (const float*)d_in[0];
  const float* alpha  = (const float*)d_in[1];
  const float* ngamma = (const float*)d_in[2];
  const float* nbeta  = (const float*)d_in[3];
  const float* w1     = (const float*)d_in[4];
  const float* b1     = (const float*)d_in[5];
  const float* w2     = (const float*)d_in[6];
  const float* b2     = (const float*)d_in[7];
  const float* fgamma = (const float*)d_in[8];
  const float* fbeta  = (const float*)d_in[9];
  float* out = (float*)d_out;

  char* ws = (char*)d_ws;
  size_t off = 0;
  float*    phi1 = (float*)(ws + off);            off += (size_t)M_ * K_ * 4;   // 64 MB
  size_t    xoff = off;
  _Float16* xbuf = (_Float16*)(ws + off);         off += (size_t)M_ * F_ * 2;   // 64 MB
  _Float16* hbuf = (_Float16*)(ws + off);         off += (size_t)M_ * H_ * 2;   // 64 MB
  _Float16* w1h  = (_Float16*)(ws + off);         off += (size_t)H_ * F_ * 2;   //  8 MB
  _Float16* w2h  = (_Float16*)(ws + off);         off += (size_t)K_ * H_ * 2;   //  4 MB
  float*    part = (float*)(ws + off);            off += (size_t)B_ * NT * K_ * 4; // 2 MB
  float*    zbuf = (float*)(ws + xoff);           // aliases xbuf (dead after GEMM1)

  f32_to_f16_kernel<<<(H_ * F_ / 4 + 255) / 256, 256, 0, stream>>>(w1, w1h, H_ * F_ / 4);
  f32_to_f16_kernel<<<(K_ * H_ / 4 + 255) / 256, 256, 0, stream>>>(w2, w2h, K_ * H_ / 4);
  scan_partial_kernel<<<B_ * NT, 256, 0, stream>>>(phi, part);
  scan_tiles_kernel<<<B_ * K_ / 256, 256, 0, stream>>>(part);
  phase_stage1_kernel<<<B_ * NT, 256, 0, stream>>>(phi, part, alpha, ngamma, nbeta, phi1, xbuf);
  wmma_gemm_kernel<0><<<dim3(H_ / 128, M_ / 128), 256, 0, stream>>>(xbuf, w1h, b1, nullptr, hbuf, nullptr);
  wmma_gemm_kernel<1><<<dim3(K_ / 128, M_ / 128), 256, 0, stream>>>(hbuf, w2h, b2, phi1, nullptr, zbuf);
  phase_final_kernel<<<M_, 256, 0, stream>>>(zbuf, fgamma, fbeta, out);
}
